// PoetryModel_29515015258593
// MI455X (gfx1250) — matrix operand order repaired and checked by reading, not compiled, not run on hardware
//
#include <hip/hip_runtime.h>

// ---------------------------------------------------------------------------
// Types / constants
// ---------------------------------------------------------------------------
typedef __attribute__((ext_vector_type(16))) _Float16 v16h;
typedef __attribute__((ext_vector_type(8)))  _Float16 v8h;
typedef __attribute__((ext_vector_type(8)))  float    v8f;

#define WMMA_F32_F16(a, b, c) \
  __builtin_amdgcn_wmma_f32_16x16x32_f16(false, (a), false, (b), (short)0, (c), false, false)

constexpr int Bn  = 4;
constexpr int Tn  = 1024;
constexpr int En  = 512;
constexpr int Hn  = 8;
constexpr int HDn = 64;
constexpr int NBn = 5;
constexpr int Vn  = 32000;
constexpr int Mrows = Bn * Tn;      // 4096

// ---------------------------------------------------------------------------
// Fragment loaders (wave32 WMMA f16 16x16x32)
//
// A-matrix 16x32 layout (ISA 7.12.2): lanes 0-15 = rows, per-lane halves are
// K = {kb..kb+7, kb+16..kb+23}, kb = (lane>=16) ? 8 : 0.
// B-matrix 32x16 layout (same K split as A): lanes 0-15 hold K=0..7,16..23,
// lanes 16-31 hold K=8..15,24..31; column index = lane&15.
// Both loaders expect the K dimension contiguous in memory (row-major A,
// pre-transposed [N,K] weights for B).
// ---------------------------------------------------------------------------
__device__ inline v16h load_fragA(const _Float16* base, int ld, int lane) {
  const _Float16* p = base + (size_t)(lane & 15) * ld + ((lane >> 4) << 3);
  v8h lo = *(const v8h*)p;          // K = kb .. kb+7
  v8h hi = *(const v8h*)(p + 16);   // K = kb+16 .. kb+23
  v16h f;
#pragma unroll
  for (int e = 0; e < 8; ++e) { f[e] = lo[e]; f[e + 8] = hi[e]; }
  return f;
}

__device__ inline v16h load_fragB(const _Float16* base, int ld, int lane) {
  const _Float16* p = base + (size_t)(lane & 15) * ld + ((lane >> 4) << 3);
  v8h lo = *(const v8h*)p;          // K = kb .. kb+7
  v8h hi = *(const v8h*)(p + 16);   // K = kb+16 .. kb+23
  v16h f;
#pragma unroll
  for (int e = 0; e < 8; ++e) { f[e] = lo[e]; f[e + 8] = hi[e]; }
  return f;
}

__device__ inline float rmax16(float v) {
  v = fmaxf(v, __shfl_xor(v, 1));
  v = fmaxf(v, __shfl_xor(v, 2));
  v = fmaxf(v, __shfl_xor(v, 4));
  v = fmaxf(v, __shfl_xor(v, 8));
  return v;
}
__device__ inline float rsum16(float v) {
  v += __shfl_xor(v, 1);
  v += __shfl_xor(v, 2);
  v += __shfl_xor(v, 4);
  v += __shfl_xor(v, 8);
  return v;
}

// ---------------------------------------------------------------------------
// Weight convert: W[batch, K, N] (f32) -> out[batch, N, K] (f16), i.e. K made
// contiguous so GEMM B-fragment loads are 16B vector loads.
// ---------------------------------------------------------------------------
__global__ void __launch_bounds__(256) w2h_kernel(const float* __restrict__ W,
                                                  _Float16* __restrict__ out,
                                                  int batch, int K, int N) {
  size_t total = (size_t)batch * K * N;
  for (size_t i = (size_t)blockIdx.x * blockDim.x + threadIdx.x; i < total;
       i += (size_t)gridDim.x * blockDim.x) {
    size_t kk  = i % (size_t)K;
    size_t rem = i / (size_t)K;
    size_t n   = rem % (size_t)N;
    size_t bb  = rem / (size_t)N;
    out[i] = (_Float16)W[(bb * K + kk) * N + n];
  }
}

// ---------------------------------------------------------------------------
// Per-head transpose of q: qT[b,h][o][t] = q16[(b,t)][h*64+o].
// Makes the attention PV B-fragments contiguous along K (=s) so they load as
// two 16B vectors instead of a 16-element scalar gather.
// ---------------------------------------------------------------------------
__global__ void __launch_bounds__(256) qtr_kernel(const _Float16* __restrict__ q16,
                                                  _Float16* __restrict__ qT) {
  size_t total = (size_t)Mrows * En;           // 2M elements
  for (size_t i = (size_t)blockIdx.x * blockDim.x + threadIdx.x; i < total;
       i += (size_t)gridDim.x * blockDim.x) {
    int t    = (int)(i & (Tn - 1));
    size_t r = i >> 10;
    int o    = (int)(r & (HDn - 1));
    int h    = (int)((r >> 6) & (Hn - 1));
    int b    = (int)(r >> 9);
    qT[i] = q16[((size_t)(b * Tn + t) << 9) + h * HDn + o];
  }
}

// ---------------------------------------------------------------------------
// Embedding: h[b,t,:] = tok_emb[x[b,t]] + pos_emb[t]
// ---------------------------------------------------------------------------
__global__ void __launch_bounds__(256) embed_kernel(const int* __restrict__ x,
                                                    const float* __restrict__ tok,
                                                    const float* __restrict__ pos,
                                                    float* __restrict__ h) {
  size_t i = (size_t)blockIdx.x * blockDim.x + threadIdx.x;
  size_t total = (size_t)Mrows * En;
  if (i >= total) return;
  size_t rt = i >> 9;             // row (b*T + t), E == 512
  int e = (int)(i & 511);
  int t = (int)(rt & (Tn - 1));
  h[i] = tok[(size_t)x[rt] * En + e] + pos[(size_t)t * En + e];
}

// ---------------------------------------------------------------------------
// LayerNorm: one wave per row of 512. Emits f16 normalized row, optionally an
// f16 copy of the raw input (needed as GEMM A operand for the residual GEMM).
// ---------------------------------------------------------------------------
__global__ void __launch_bounds__(256) ln_kernel(const float* __restrict__ h,
                                                 const float* __restrict__ g,
                                                 const float* __restrict__ bb,
                                                 _Float16* __restrict__ hn16,
                                                 _Float16* __restrict__ h16,
                                                 int rows) {
  int wid = threadIdx.x >> 5, lane = threadIdx.x & 31;
  int row = blockIdx.x * 8 + wid;
  if (row >= rows) return;
  const float* x = h + (size_t)row * En;
  float v[16];
  float s = 0.f;
#pragma unroll
  for (int i = 0; i < 16; ++i) { v[i] = x[i * 32 + lane]; s += v[i]; }
#pragma unroll
  for (int off = 1; off < 32; off <<= 1) s += __shfl_xor(s, off);
  float mu = s * (1.0f / En);
  float vs = 0.f;
#pragma unroll
  for (int i = 0; i < 16; ++i) { float d = v[i] - mu; vs += d * d; }
#pragma unroll
  for (int off = 1; off < 32; off <<= 1) vs += __shfl_xor(vs, off);
  float rstd = rsqrtf(vs * (1.0f / En) + 1e-5f);
#pragma unroll
  for (int i = 0; i < 16; ++i) {
    int c = i * 32 + lane;
    float y = (v[i] - mu) * rstd * g[c] + bb[c];
    hn16[(size_t)row * En + c] = (_Float16)y;
    if (h16) h16[(size_t)row * En + c] = (_Float16)v[i];
  }
}

// ---------------------------------------------------------------------------
// Generic WMMA GEMM:  C[M,N] = A16[M,K] @ BT16[N,K]^T  (+bias, relu, +resid)
// 256 threads = 8 waves arranged 4(M) x 2(N).
// Wave tile: 32 x (WN*16).  Block tile: 128 x (WN*32).
//   WN=2 -> 128x64  (2x2 WMMA/wave, for the E-wide GEMMs, more blocks)
//   WN=4 -> 128x128 (2x4 WMMA/wave, 8 WMMA : 6 frag loads, for vocab proj)
// M%128==0, N%(WN*32)==0, K%32==0 for all call sites here.
// ---------------------------------------------------------------------------
template <int WN>
__global__ void __launch_bounds__(256) gemm_kernel(const _Float16* __restrict__ A,
                                                   const _Float16* __restrict__ BT,
                                                   int M, int N, int K,
                                                   const float* __restrict__ bias,
                                                   const float* __restrict__ resid,
                                                   int relu,
                                                   float* __restrict__ outF,
                                                   _Float16* __restrict__ outH) {
  int wid = threadIdx.x >> 5, lane = threadIdx.x & 31;
  int wm = wid & 3, wn = wid >> 2;
  int m0 = blockIdx.y * 128 + wm * 32;
  int n0 = blockIdx.x * (WN * 32) + wn * (WN * 16);
  const _Float16* Ap0 = A + (size_t)m0 * K;
  const _Float16* Ap1 = A + (size_t)(m0 + 16) * K;
  const _Float16* Bp[WN];
#pragma unroll
  for (int j = 0; j < WN; ++j) Bp[j] = BT + (size_t)(n0 + j * 16) * K;

  v8f acc0[WN], acc1[WN];
#pragma unroll
  for (int j = 0; j < WN; ++j) { acc0[j] = (v8f){}; acc1[j] = (v8f){}; }

  for (int k0 = 0; k0 < K; k0 += 32) {
    if (k0 + 64 < K) {
      __builtin_prefetch(Ap0 + k0 + 64, 0, 1);
      __builtin_prefetch(Bp[0] + k0 + 64, 0, 1);
    }
    v16h a0 = load_fragA(Ap0 + k0, K, lane);
    v16h a1 = load_fragA(Ap1 + k0, K, lane);
#pragma unroll
    for (int j = 0; j < WN; ++j) {
      v16h b = load_fragB(Bp[j] + k0, K, lane);
      acc0[j] = WMMA_F32_F16(a0, b, acc0[j]);
      acc1[j] = WMMA_F32_F16(a1, b, acc1[j]);
    }
  }

  int hoff = (lane >> 4) << 3;   // C layout: lanes 16-31 own rows M+8
  int ncol = lane & 15;
  auto emit = [&](v8f c, int mi, int nj) {
    int n = n0 + nj * 16 + ncol;
    float bv = bias ? bias[n] : 0.f;
#pragma unroll
    for (int r = 0; r < 8; ++r) {
      int m = m0 + mi * 16 + hoff + r;
      float v = c[r] + bv;
      if (relu)  v = fmaxf(v, 0.f);
      if (resid) v += resid[(size_t)m * N + n];
      if (outF)  outF[(size_t)m * N + n] = v;
      if (outH)  outH[(size_t)m * N + n] = (_Float16)v;
    }
  };
#pragma unroll
  for (int j = 0; j < WN; ++j) { emit(acc0[j], 0, j); emit(acc1[j], 1, j); }
}

// ---------------------------------------------------------------------------
// Flash attention with the reference quirks:
//   scores[t,s] = (k[t] . q[s]) * E^-0.5, causal, V == q.
// One wave handles 16 queries of one (b,h); streams keys 32 at a time with
// online softmax. P (C-layout f32) -> A-layout f16 via a per-wave LDS tile
// with an explicit s_wait_dscnt between the transpose store and reload.
// V fragments come from the pre-transposed qT[b,h][o][t] so every fragment
// load in the loop is two 16B vector loads.
// ---------------------------------------------------------------------------
__global__ void __launch_bounds__(128) attn_kernel(const _Float16* __restrict__ k16,
                                                   const _Float16* __restrict__ q16,
                                                   const _Float16* __restrict__ qT,
                                                   float* __restrict__ mha) {
  __shared__ __align__(16) _Float16 ldsP[4][16 * 32];
  int wid = threadIdx.x >> 5, lane = threadIdx.x & 31;
  int idx = blockIdx.x * 4 + wid;        // 0 .. B*H*(T/16)-1 = 2047
  int bh = idx >> 6;                     // T/16 = 64 query tiles per (b,h)
  int qt = idx & 63;
  int b = bh >> 3, h = bh & 7;
  int t0 = qt * 16;
  const _Float16* khead  = k16 + (size_t)b * Tn * En + h * HDn;
  const _Float16* qhead  = q16 + (size_t)b * Tn * En + h * HDn;
  const _Float16* qThead = qT + ((size_t)(b * Hn + h) * HDn) * Tn;

  // "query side" = k rows (A fragments), kept resident across the key loop
  v16h a_k0 = load_fragA(khead + (size_t)t0 * En + 0,  En, lane);
  v16h a_k1 = load_fragA(khead + (size_t)t0 * En + 32, En, lane);

  float m_run[8], l_run[8];
#pragma unroll
  for (int r = 0; r < 8; ++r) { m_run[r] = -1e30f; l_run[r] = 0.f; }
  v8f o0 = {}, o1 = {}, o2 = {}, o3 = {};
  const float scale = 0.04419417382415922f;   // 512^-0.5 (C = E)
  int hoff = (lane >> 4) << 3;
  int ncol = lane & 15;
  int nsteps = (t0 + 15) / 32 + 1;

  for (int j = 0; j < nsteps; ++j) {
    int s0 = j * 32;
    v8f slo = {}, shi = {};
    {
      v16h bq;
      bq = load_fragB(qhead + (size_t)s0 * En + 0,  En, lane);
      slo = WMMA_F32_F16(a_k0, bq, slo);
      bq = load_fragB(qhead + (size_t)s0 * En + 32, En, lane);
      slo = WMMA_F32_F16(a_k1, bq, slo);
      bq = load_fragB(qhead + (size_t)(s0 + 16) * En + 0,  En, lane);
      shi = WMMA_F32_F16(a_k0, bq, shi);
      bq = load_fragB(qhead + (size_t)(s0 + 16) * En + 32, En, lane);
      shi = WMMA_F32_F16(a_k1, bq, shi);
    }
#pragma unroll
    for (int r = 0; r < 8; ++r) { slo[r] *= scale; shi[r] *= scale; }
    if (j == nsteps - 1) {                  // only the diagonal step needs mask
#pragma unroll
      for (int r = 0; r < 8; ++r) {
        int mg = t0 + hoff + r;
        if (s0 + ncol      > mg) slo[r] = -1e30f;
        if (s0 + 16 + ncol > mg) shi[r] = -1e30f;
      }
    }
#pragma unroll
    for (int r = 0; r < 8; ++r) {
      float mx = rmax16(fmaxf(slo[r], shi[r]));
      float nm = fmaxf(m_run[r], mx);
      float alpha = __expf(m_run[r] - nm);
      float plo = __expf(slo[r] - nm);
      float phi = __expf(shi[r] - nm);
      float rs = rsum16(plo + phi);
      m_run[r] = nm;
      l_run[r] = l_run[r] * alpha + rs;
      o0[r] *= alpha; o1[r] *= alpha; o2[r] *= alpha; o3[r] *= alpha;
      ldsP[wid][(hoff + r) * 32 + ncol]      = (_Float16)plo;
      ldsP[wid][(hoff + r) * 32 + 16 + ncol] = (_Float16)phi;
    }
    asm volatile("s_wait_dscnt 0" ::: "memory");
    v16h a_p = load_fragA(&ldsP[wid][0], 32, lane);
    v16h bv;
    bv = load_fragB(qThead + (size_t)(0)  * Tn + s0, Tn, lane);
    o0 = WMMA_F32_F16(a_p, bv, o0);
    bv = load_fragB(qThead + (size_t)(16) * Tn + s0, Tn, lane);
    o1 = WMMA_F32_F16(a_p, bv, o1);
    bv = load_fragB(qThead + (size_t)(32) * Tn + s0, Tn, lane);
    o2 = WMMA_F32_F16(a_p, bv, o2);
    bv = load_fragB(qThead + (size_t)(48) * Tn + s0, Tn, lane);
    o3 = WMMA_F32_F16(a_p, bv, o3);
  }

  // mha[b, t, h*64 + o] (f32), normalized by row sums
  float* orow = mha + (size_t)(b * Tn + t0) * En + h * HDn;
#pragma unroll
  for (int r = 0; r < 8; ++r) {
    float inv = 1.f / l_run[r];
    size_t base = (size_t)(hoff + r) * En;
    orow[base + 0  + ncol] = o0[r] * inv;
    orow[base + 16 + ncol] = o1[r] * inv;
    orow[base + 32 + ncol] = o2[r] * inv;
    orow[base + 48 + ncol] = o3[r] * inv;
  }
}

// ---------------------------------------------------------------------------
// Host orchestration
// ---------------------------------------------------------------------------
extern "C" void kernel_launch(void* const* d_in, const int* in_sizes, int n_in,
                              void* d_out, int out_size, void* d_ws, size_t ws_size,
                              hipStream_t stream) {
  (void)in_sizes; (void)n_in; (void)out_size; (void)ws_size;
  const int*   x      = (const int*)  d_in[0];
  const float* tok    = (const float*)d_in[1];
  const float* pos    = (const float*)d_in[2];
  const float* Wk     = (const float*)d_in[3];   // (NB,H,E,HD)
  const float* Wq     = (const float*)d_in[4];
  const float* Wres   = (const float*)d_in[5];   // (NB,E,E)
  const float* ln1_g  = (const float*)d_in[6];
  const float* ln1_b  = (const float*)d_in[7];
  const float* mlp_W  = (const float*)d_in[8];   // (NB,E,E)
  const float* mlp_b  = (const float*)d_in[9];
  const float* ln2_g  = (const float*)d_in[10];
  const float* ln2_b  = (const float*)d_in[11];
  const float* lnf_g  = (const float*)d_in[12];
  const float* lnf_b  = (const float*)d_in[13];
  const float* projW  = (const float*)d_in[14];  // (E,VOCAB)
  const float* projB  = (const float*)d_in[15];
  float* logits = (float*)d_out;

  char* wp = (char*)d_ws;
  auto carve = [&](size_t bytes) -> void* {
    void* r = (void*)wp;
    wp += (bytes + 255) & ~(size_t)255;
    return r;
  };
  float*    h     = (float*)   carve((size_t)Mrows * En * 4);
  float*    mha   = (float*)   carve((size_t)Mrows * En * 4);
  _Float16* hn16  = (_Float16*)carve((size_t)Mrows * En * 2);
  _Float16* h16   = (_Float16*)carve((size_t)Mrows * En * 2);
  _Float16* k16   = (_Float16*)carve((size_t)Mrows * En * 2);
  _Float16* q16   = (_Float16*)carve((size_t)Mrows * En * 2);
  _Float16* qT16  = (_Float16*)carve((size_t)Mrows * En * 2);
  _Float16* WkT   = (_Float16*)carve((size_t)NBn * En * En * 2);
  _Float16* WqT   = (_Float16*)carve((size_t)NBn * En * En * 2);
  _Float16* WresT = (_Float16*)carve((size_t)NBn * En * En * 2);
  _Float16* mlpT  = (_Float16*)carve((size_t)NBn * En * En * 2);
  _Float16* projT = (_Float16*)carve((size_t)En * Vn * 2);

  // Weight conversion + transpose to [N, K] f16 (cheap, L2-resident afterwards)
  w2h_kernel<<<2048, 256, 0, stream>>>(Wk,    WkT,   NBn * Hn, En, HDn);
  w2h_kernel<<<2048, 256, 0, stream>>>(Wq,    WqT,   NBn * Hn, En, HDn);
  w2h_kernel<<<2048, 256, 0, stream>>>(Wres,  WresT, NBn,      En, En);
  w2h_kernel<<<2048, 256, 0, stream>>>(mlp_W, mlpT,  NBn,      En, En);
  w2h_kernel<<<4096, 256, 0, stream>>>(projW, projT, 1,        En, Vn);

  embed_kernel<<<(Mrows * En + 255) / 256, 256, 0, stream>>>(x, tok, pos, h);

  dim3 g512(En / 64, Mrows / 128);     // (8, 32), WN=2 tiles
  for (int nb = 0; nb < NBn; ++nb) {
    size_t wOff = (size_t)nb * En * En;
    // LN1 -> hn16 (+ f16 copy of h for the residual GEMM's A operand)
    ln_kernel<<<512, 256, 0, stream>>>(h, ln1_g + nb * En, ln1_b + nb * En,
                                       hn16, h16, Mrows);
    // k / q projections: [4096,512] @ [512,512]
    gemm_kernel<2><<<g512, 256, 0, stream>>>(hn16, WkT + wOff, Mrows, En, En,
                                             nullptr, nullptr, 0, nullptr, k16);
    gemm_kernel<2><<<g512, 256, 0, stream>>>(hn16, WqT + wOff, Mrows, En, En,
                                             nullptr, nullptr, 0, nullptr, q16);
    // per-head transpose of q for the PV matmul's B operand
    qtr_kernel<<<2048, 256, 0, stream>>>(q16, qT16);
    // attention (v == q), writes mha f32
    attn_kernel<<<512, 128, 0, stream>>>(k16, q16, qT16, mha);
    // h = h @ Wres + mha
    gemm_kernel<2><<<g512, 256, 0, stream>>>(h16, WresT + wOff, Mrows, En, En,
                                             nullptr, mha, 0, h, nullptr);
    // LN2 -> hn16
    ln_kernel<<<512, 256, 0, stream>>>(h, ln2_g + nb * En, ln2_b + nb * En,
                                       hn16, nullptr, Mrows);
    // h = h + relu(hn2 @ mlp_W + mlp_b)
    gemm_kernel<2><<<g512, 256, 0, stream>>>(hn16, mlpT + wOff, Mrows, En, En,
                                             mlp_b + nb * En, h, 1, h, nullptr);
  }

  // final LN + vocab projection into d_out (wide 128x128 tiles, 8 WMMA/step)
  ln_kernel<<<512, 256, 0, stream>>>(h, lnf_g, lnf_b, hn16, nullptr, Mrows);
  dim3 gproj(Vn / 128, Mrows / 128);   // (250, 32), WN=4 tiles
  gemm_kernel<4><<<gproj, 256, 0, stream>>>(hn16, projT, Mrows, Vn, En,
                                            projB, nullptr, 0, logits, nullptr);
}
